// Encoder_Model_63290638074676
// MI455X (gfx1250) — matrix operand order repaired and checked by reading, not compiled, not run on hardware
//
#include <hip/hip_runtime.h>
#include <hip/hip_bf16.h>
#include <math.h>

// ---------------- constants (match reference) ----------------
#define CN 30000      // N entities
#define CR 2000       // R relations
#define CT 300000     // T triples
#define CD 128        // D
#define CE 600000     // E edges
#define CB 1024       // B pairs
#define CPX 64        // proxies
#define CDD 384       // (DEPTH+1)*D
#define CJD 768       // joint dim
#define CGAMMA 3.0f
#define CEPS 1e-12f

typedef __bf16 bf16_t;
typedef bf16_t v16bf __attribute__((ext_vector_type(16)));
typedef float  v8f   __attribute__((ext_vector_type(8)));

__device__ __forceinline__ bf16_t f2bf(float x) {
    unsigned u = __float_as_uint(x);
    u += 0x7FFFu + ((u >> 16) & 1u);           // round-to-nearest-even
    unsigned short s = (unsigned short)(u >> 16);
    bf16_t r; __builtin_memcpy(&r, &s, 2);
    return r;
}

// monotone float<->uint encoding for atomic max over signed floats
__device__ __forceinline__ unsigned fenc(float f) {
    unsigned u = __float_as_uint(f);
    return (u & 0x80000000u) ? ~u : (u | 0x80000000u);
}
__device__ __forceinline__ float fdec(unsigned u) {
    unsigned v = (u & 0x80000000u) ? (u & 0x7FFFFFFFu) : ~u;
    return __uint_as_float(v);
}

__device__ __forceinline__ float wave_sum(float s) {
#pragma unroll
    for (int o = 16; o > 0; o >>= 1) s += __shfl_xor(s, o);
    return s;
}

// ---------------- utility kernels ----------------
__global__ void k_zero_f32(float* p, size_t n) {
    size_t i = (size_t)blockIdx.x * blockDim.x + threadIdx.x;
    size_t st = (size_t)gridDim.x * blockDim.x;
    for (; i < n; i += st) p[i] = 0.0f;
}

__global__ void k_to_bf16(const float* in, bf16_t* out, size_t n) {
    size_t i = (size_t)blockIdx.x * blockDim.x + threadIdx.x;
    size_t st = (size_t)gridDim.x * blockDim.x;
    for (; i < n; i += st) out[i] = f2bf(in[i]);
}

// transpose K x N f32 -> N x K bf16 (so GEMM B-operand reads are K-contiguous)
__global__ void k_transpose_to_bf16(const float* in, bf16_t* out, int Kd, int Nd) {
    int idx = blockIdx.x * blockDim.x + threadIdx.x;
    if (idx >= Kd * Nd) return;
    int n = idx / Kd, k = idx - n * Kd;
    out[idx] = f2bf(in[(size_t)k * Nd + n]);
}

// ---------------- segment-mean aggregation ----------------
__global__ void k_count_rows(const int* rows, int E, float* cnt) {
    int i = blockIdx.x * blockDim.x + threadIdx.x;
    if (i < E) atomicAdd(&cnt[rows[i]], 1.0f);
}

__global__ void k_avg_scatter(const int* rows, const int* cols, int E,
                              const float* emb, const float* cnt, float* outf) {
    int e = (blockIdx.x * blockDim.x + threadIdx.x) >> 5;
    int lane = threadIdx.x & 31;
    if (e >= E) return;
    int r = rows[e], c = cols[e];
    float w = 1.0f / cnt[r];
    const float* src = emb + (size_t)c * CD;
    float* dst = outf + (size_t)r * CD;
    for (int d = lane; d < CD; d += 32) atomicAdd(&dst[d], w * src[d]);
}

// rels_sum scatter: rels[r0[t]] += r_val[t] * rel_emb[r1[t]]
__global__ void k_rels_scatter(const int* r0, const int* r1, const float* rval,
                               int Tn, const float* rel_emb, float* rels) {
    int t = (blockIdx.x * blockDim.x + threadIdx.x) >> 5;
    int lane = threadIdx.x & 31;
    if (t >= Tn) return;
    float rv = rval[t];
    const float* src = rel_emb + (size_t)r1[t] * CD;
    float* dst = rels + (size_t)r0[t] * CD;
    for (int d = lane; d < CD; d += 32) atomicAdd(&dst[d], rv * src[d]);
}

// generic row-wise l2 normalize (wave per row), in-place safe
__global__ void k_l2norm_rows(const float* in, float* out, int rows, int cols) {
    int r = (blockIdx.x * blockDim.x + threadIdx.x) >> 5;
    int lane = threadIdx.x & 31;
    if (r >= rows) return;
    const float* p = in + (size_t)r * cols;
    float ss = 0.0f;
    for (int d = lane; d < cols; d += 32) { float v = p[d]; ss += v * v; }
    ss = wave_sum(ss);
    float inv = 1.0f / fmaxf(sqrtf(ss), CEPS);
    float* q = out + (size_t)r * cols;
    for (int d = lane; d < cols; d += 32) q[d] = p[d] * inv;
}

// row sum of squares (wave per row)
__global__ void k_rowsq(const float* m, float* out, int rows, int cols) {
    int r = (blockIdx.x * blockDim.x + threadIdx.x) >> 5;
    int lane = threadIdx.x & 31;
    if (r >= rows) return;
    const float* p = m + (size_t)r * cols;
    float ss = 0.0f;
    for (int d = lane; d < cols; d += 32) { float v = p[d]; ss += v * v; }
    ss = wave_sum(ss);
    if (lane == 0) out[r] = ss;
}

// elementwise tanh: write into feats buffer and into joint[:, jbase:jbase+128]
__global__ void k_tanh_to(const float* in, float* feats, float* joint, int jbase, size_t total) {
    size_t i = (size_t)blockIdx.x * blockDim.x + threadIdx.x;
    if (i >= total) return;
    size_t n = i / CD;
    int d = (int)(i - n * CD);
    float v = tanhf(in[i]);
    feats[i] = v;
    joint[n * CJD + jbase + d] = v;
}

// ---------------- GAT layer kernels ----------------
// att[t] = rels[t] . attn ; atomicMax amax[row[t]]
__global__ void k_att_max(const float* rels, const float* attn, const int* rowi,
                          int Tn, float* att, unsigned* amax) {
    int t = (blockIdx.x * blockDim.x + threadIdx.x) >> 5;
    int lane = threadIdx.x & 31;
    if (t >= Tn) return;
    const float* rp = rels + (size_t)t * CD;
    float s = 0.0f;
    for (int d = lane; d < CD; d += 32) s += rp[d] * attn[d];
    s = wave_sum(s);
    if (lane == 0) {
        att[t] = s;
        atomicMax(&amax[rowi[t]], fenc(s));
    }
}

// e[t] = exp(att-amax[row]); denom[row] += e (att overwritten with e)
__global__ void k_exp_denom(float* att, const unsigned* amax, const int* rowi,
                            float* denom, int Tn) {
    int t = blockIdx.x * blockDim.x + threadIdx.x;
    if (t >= Tn) return;
    int r = rowi[t];
    float e = expf(att[t] - fdec(amax[r]));
    att[t] = e;
    atomicAdd(&denom[r], e);
}

// newf[row] += w * (feats[col] - 2*(feats[col].rels[t])*rels[t]),  w = e/denom[row]
__global__ void k_layer_scatter(const float* feats, const float* rels, const float* e,
                                const float* denom, const int* rowi, const int* coli,
                                int Tn, float* newf) {
    int t = (blockIdx.x * blockDim.x + threadIdx.x) >> 5;
    int lane = threadIdx.x & 31;
    if (t >= Tn) return;
    int r = rowi[t], c = coli[t];
    float w = e[t] / denom[r];
    const float* fp = feats + (size_t)c * CD;
    const float* rp = rels + (size_t)t * CD;
    float dot = 0.0f;
    for (int d = lane; d < CD; d += 32) dot += fp[d] * rp[d];
    dot = wave_sum(dot);
    float* dst = newf + (size_t)r * CD;
    for (int d = lane; d < CD; d += 32)
        atomicAdd(&dst[d], w * (fp[d] - 2.0f * dot * rp[d]));
}

// ---------------- proxy attention + pf ----------------
// block (64 threads) per row: pa = softmax(l2norm(o) @ pnorm^T); pf = o - pa@proxy
__global__ void k_proxy(const float* joint, int jbase, const float* proxy,
                        const float* pnorm, float* pf, bf16_t* pfbf) {
    __shared__ float o[CDD];
    __shared__ float pa[CPX];
    __shared__ float red[CPX];
    int n = blockIdx.x, t = threadIdx.x;
    const float* orow = joint + (size_t)n * CJD + jbase;
    float ss = 0.0f;
    for (int d = t; d < CDD; d += CPX) { float v = orow[d]; o[d] = v; ss += v * v; }
    red[t] = ss; __syncthreads();
    for (int s = 32; s > 0; s >>= 1) { if (t < s) red[t] += red[t + s]; __syncthreads(); }
    float inv = 1.0f / fmaxf(sqrtf(red[0]), CEPS);
    __syncthreads();
    const float* pr = pnorm + (size_t)t * CDD;
    float dot = 0.0f;
    for (int d = 0; d < CDD; ++d) dot += o[d] * pr[d];
    dot *= inv;
    red[t] = dot; __syncthreads();
    float m = red[0];
    for (int j = 1; j < CPX; ++j) m = fmaxf(m, red[j]);
    float ev = expf(dot - m);
    pa[t] = ev; __syncthreads();
    float sum = 0.0f;
    for (int j = 0; j < CPX; ++j) sum += pa[j];
    float w = ev / sum;
    __syncthreads();
    pa[t] = w; __syncthreads();
    for (int d = t; d < CDD; d += CPX) {
        float s2 = 0.0f;
        for (int j = 0; j < CPX; ++j) s2 += pa[j] * proxy[(size_t)j * CDD + d];
        float v = o[d] - s2;
        pf[(size_t)n * CDD + d] = v;
        pfbf[(size_t)n * CDD + d] = f2bf(v);
    }
}

// joint_slice = g*out + (1-g)*pf,  g = sigmoid(gkout + bias)
__global__ void k_combine(const float* gk, const float* bias, const float* pf,
                          float* joint, int jbase, size_t total) {
    size_t i = (size_t)blockIdx.x * blockDim.x + threadIdx.x;
    if (i >= total) return;
    size_t n = i / CDD;
    int d = (int)(i - n * CDD);
    float g = 1.0f / (1.0f + expf(-(gk[i] + bias[d])));
    size_t ji = n * CJD + jbase + d;
    float o = joint[ji];
    joint[ji] = g * o + (1.0f - g) * pf[i];
}

// ---------------- WMMA bf16 GEMM ----------------
// C[M,N] = A[M,K] * B[N,K]^T  (both operands row-major with K contiguous)
// Wave tile: 16(M) x 64(N) -> 4 f32 accumulators, A fragment reused 4x.
// Software-pipelined K-loop: fragments for k+32 are loaded (double-buffered in
// registers) before the 4 v_wmma_f32_16x16x32_bf16 of step k issue, so the
// WMMAs wait only on loads issued one iteration earlier (loads stay in flight).
__device__ __forceinline__ v16bf load_a_frag(const bf16_t* __restrict__ Ap, int k0, int hi) {
    v16bf a;
#pragma unroll
    for (int e = 0; e < 8; ++e) a[e] = Ap[k0 + hi * 8 + e];
#pragma unroll
    for (int e = 0; e < 8; ++e) a[8 + e] = Ap[k0 + 16 + hi * 8 + e];
    return a;
}
__device__ __forceinline__ v16bf load_b_frag(const bf16_t* __restrict__ Bp, int k0, int hi) {
    v16bf b;
#pragma unroll
    for (int e = 0; e < 16; ++e) b[e] = Bp[k0 + (e < 8 ? hi * 8 + e : 16 + hi * 8 + (e - 8))];
    return b;
}

__global__ __launch_bounds__(128) void k_gemm_bf16_tn(const bf16_t* __restrict__ A,
                                                      const bf16_t* __restrict__ B,
                                                      float* __restrict__ C,
                                                      int M, int N, int K) {
    int lane = threadIdx.x & 31;
    int wave = threadIdx.x >> 5;
    int ntiles = N >> 4;                       // N is a multiple of 16
    int ntile0 = (blockIdx.x * 4 + wave) * 4;  // first of 4 n-tiles for this wave
    if (ntile0 >= ntiles) return;
    int tm = blockIdx.y;
    int hi = lane >> 4;                        // half-wave select
    int l15 = lane & 15;
    int mrow = tm * 16 + l15;

    const bf16_t* Ap = A + (size_t)mrow * K;
    int nt[4];
    const bf16_t* Bp[4];
#pragma unroll
    for (int j = 0; j < 4; ++j) {
        int t = ntile0 + j;
        nt[j] = (t < ntiles) ? t : (ntiles - 1); // clamp; store guarded below
        Bp[j] = B + (size_t)(nt[j] * 16 + l15) * K;
    }

    v8f acc[4];
#pragma unroll
    for (int j = 0; j < 4; ++j)
#pragma unroll
        for (int v = 0; v < 8; ++v) acc[j][v] = 0.0f;

    // prologue: load fragments for k = 0
    v16bf a_cur = load_a_frag(Ap, 0, hi);
    v16bf b_cur[4];
#pragma unroll
    for (int j = 0; j < 4; ++j) b_cur[j] = load_b_frag(Bp[j], 0, hi);

    int k0 = 0;
    for (; k0 + 32 < K; k0 += 32) {
        int kn = k0 + 32;
        if (kn + 32 < K) {                      // prefetch one step further ahead
#pragma unroll
            for (int j = 0; j < 4; ++j) __builtin_prefetch(Bp[j] + kn + 32, 0, 1);
        }
        // issue next-step loads before consuming current fragments
        v16bf a_nxt = load_a_frag(Ap, kn, hi);
        v16bf b_nxt[4];
#pragma unroll
        for (int j = 0; j < 4; ++j) b_nxt[j] = load_b_frag(Bp[j], kn, hi);
#pragma unroll
        for (int j = 0; j < 4; ++j)
            acc[j] = __builtin_amdgcn_wmma_f32_16x16x32_bf16(false, a_cur, false, b_cur[j],
                                                             (short)0, acc[j], false, false);
        a_cur = a_nxt;
#pragma unroll
        for (int j = 0; j < 4; ++j) b_cur[j] = b_nxt[j];
    }
    // epilogue: last K-step
#pragma unroll
    for (int j = 0; j < 4; ++j)
        acc[j] = __builtin_amdgcn_wmma_f32_16x16x32_bf16(false, a_cur, false, b_cur[j],
                                                         (short)0, acc[j], false, false);

    // C layout: VGPR v -> M = tm*16 + v + 8*hi ; N = nt*16 + l15
#pragma unroll
    for (int j = 0; j < 4; ++j) {
        if (ntile0 + j >= ntiles) break;
        float* Cp = C + (size_t)(tm * 16 + 8 * hi) * N + nt[j] * 16 + l15;
#pragma unroll
        for (int v = 0; v < 8; ++v) Cp[(size_t)v * N] = acc[j][v];
    }
}

// ---------------- align loss ----------------
// block per pair: pack l/r rows of joint (bf16), pos = ||l-r||^2, asq gather
__global__ void k_build_pairs(const float* joint, const int* pairs, bf16_t* Abf,
                              float* pos, float* asq, const float* embsq) {
    __shared__ float red[128];
    int b = blockIdx.x, t = threadIdx.x;
    int l = pairs[2 * b], r = pairs[2 * b + 1];
    const float* lr = joint + (size_t)l * CJD;
    const float* rr = joint + (size_t)r * CJD;
    float ps = 0.0f;
    for (int d = t; d < CJD; d += 128) {
        float lv = lr[d], rv = rr[d];
        Abf[(size_t)b * CJD + d] = f2bf(lv);
        Abf[(size_t)(CB + b) * CJD + d] = f2bf(rv);
        float df = lv - rv;
        ps += df * df;
    }
    red[t] = ps; __syncthreads();
    for (int s = 64; s > 0; s >>= 1) { if (t < s) red[t] += red[t + s]; __syncthreads(); }
    if (t == 0) {
        pos[b] = red[0];
        asq[b] = embsq[l];
        asq[CB + b] = embsq[r];
    }
}

// block (256 threads) per row i of ab (2048 rows): standardize then logsumexp
__global__ void k_row_loss(const float* ab, const float* pos, const float* asq,
                           const float* embsq, const int* pairs, float* lse) {
    __shared__ float s1[256], s2[256];
    int i = blockIdx.x, t = threadIdx.x;
    int b = i & (CB - 1);
    int l = pairs[2 * b], r = pairs[2 * b + 1];
    float posv = pos[b];
    float aq = asq[i];
    const float* abr = ab + (size_t)i * CN;
    float sum = 0.0f, sumsq = 0.0f;
    for (int n = t; n < CN; n += 256) {
        float neg = aq + embsq[n] - 2.0f * abr[n];
        float mask = 1.0f + (n == l ? -1.0f : 0.0f) + (n == r ? -1.0f : 0.0f);
        float x = (posv - neg + CGAMMA) * mask;
        sum += x; sumsq += x * x;
    }
    s1[t] = sum; s2[t] = sumsq; __syncthreads();
    for (int s = 128; s > 0; s >>= 1) {
        if (t < s) { s1[t] += s1[t + s]; s2[t] += s2[t + s]; }
        __syncthreads();
    }
    float mu = s1[0] / (float)CN;
    float var = s2[0] / (float)CN - mu * mu;
    float inv_sd = 1.0f / fmaxf(sqrtf(fmaxf(var, 0.0f)), 1e-30f);
    __syncthreads();
    // streaming logsumexp of 20*(x-mu)/sd + 8
    float lm = -3.4e38f, ls = 0.0f;
    for (int n = t; n < CN; n += 256) {
        float neg = aq + embsq[n] - 2.0f * abr[n];
        float mask = 1.0f + (n == l ? -1.0f : 0.0f) + (n == r ? -1.0f : 0.0f);
        float x = (posv - neg + CGAMMA) * mask;
        float y = 20.0f * (x - mu) * inv_sd + 8.0f;
        if (y > lm) { ls = ls * expf(lm - y) + 1.0f; lm = y; }
        else        { ls += expf(y - lm); }
    }
    s1[t] = lm; s2[t] = ls; __syncthreads();
    if (t == 0) {
        float M = -3.4e38f;
        for (int j = 0; j < 256; ++j) M = fmaxf(M, s1[j]);
        float S = 0.0f;
        for (int j = 0; j < 256; ++j) S += s2[j] * expf(s1[j] - M);
        lse[i] = M + logf(S);
    }
}

__global__ void k_final(const float* lse, float* out) {
    __shared__ float s[256];
    int t = threadIdx.x;
    float acc = 0.0f;
    for (int b = t; b < CB; b += 256) acc += lse[b] + lse[CB + b];
    s[t] = acc; __syncthreads();
    for (int k = 128; k > 0; k >>= 1) { if (t < k) s[t] += s[t + k]; __syncthreads(); }
    if (t == 0) out[0] = s[0] / (float)CB;
}

// ---------------- host orchestration ----------------
extern "C" void kernel_launch(void* const* d_in, const int* in_sizes, int n_in,
                              void* d_out, int out_size, void* d_ws, size_t ws_size,
                              hipStream_t stream) {
    (void)in_sizes; (void)n_in; (void)out_size; (void)ws_size;
    const float* ent_emb = (const float*)d_in[0];
    const float* rel_emb = (const float*)d_in[1];
    const float* e_attn  = (const float*)d_in[2];
    const float* e_gate  = (const float*)d_in[3];
    const float* e_proxy = (const float*)d_in[4];
    const float* e_bias  = (const float*)d_in[5];
    const float* r_attn  = (const float*)d_in[6];
    const float* r_gate  = (const float*)d_in[7];
    const float* r_proxy = (const float*)d_in[8];
    const float* r_bias  = (const float*)d_in[9];
    const float* r_val   = (const float*)d_in[10];
    const int* ent_adj   = (const int*)d_in[11];   // (2, E)
    const int* rel_adj   = (const int*)d_in[12];   // (2, E)
    const int* adj       = (const int*)d_in[13];   // (2, T)
    const int* r_index   = (const int*)d_in[14];   // (2, T)
    const int* pairs     = (const int*)d_in[15];   // (B, 2)
    float* out = (float*)d_out;

    char* base = (char*)d_ws;
    size_t off = 0;
    auto alloc = [&](size_t bytes) -> void* {
        void* p = base + off;
        off = (off + bytes + 255) & ~(size_t)255;
        return p;
    };

    float*    cnt     = (float*)alloc((size_t)CN * 4);
    float*    feat_e  = (float*)alloc((size_t)CN * CD * 4);
    float*    feat_r  = (float*)alloc((size_t)CN * CD * 4);
    float*    rels    = (float*)alloc((size_t)CT * CD * 4);
    float*    att     = (float*)alloc((size_t)CT * 4);
    unsigned* amax    = (unsigned*)alloc((size_t)CN * 4);
    float*    denom   = (float*)alloc((size_t)CN * 4);
    float*    featsA  = (float*)alloc((size_t)CN * CD * 4);
    float*    featsB  = (float*)alloc((size_t)CN * CD * 4);
    float*    joint   = (float*)alloc((size_t)CN * CJD * 4);
    bf16_t*   jointbf = (bf16_t*)alloc((size_t)CN * CJD * 2);
    float*    pf      = (float*)alloc((size_t)CN * CDD * 4);
    bf16_t*   pfbf    = (bf16_t*)alloc((size_t)CN * CDD * 2);
    bf16_t*   gktbf   = (bf16_t*)alloc((size_t)CDD * CDD * 2);  // gate_k^T (N x K)
    float*    gkout   = (float*)alloc((size_t)CN * CDD * 4);
    float*    pnorm   = (float*)alloc((size_t)CPX * CDD * 4);
    bf16_t*   Abf     = (bf16_t*)alloc((size_t)2 * CB * CJD * 2);
    float*    ab      = (float*)alloc((size_t)2 * CB * CN * 4);
    float*    pos     = (float*)alloc((size_t)CB * 4);
    float*    asq     = (float*)alloc((size_t)2 * CB * 4);
    float*    embsq   = (float*)alloc((size_t)CN * 4);
    float*    lse     = (float*)alloc((size_t)2 * CB * 4);

    const int ZB = 4096;
    // ---- ent_feature = avg(ent_adj, ent_emb) ----
    k_zero_f32<<<ZB, 256, 0, stream>>>(cnt, (size_t)CN);
    k_zero_f32<<<ZB, 256, 0, stream>>>(feat_e, (size_t)CN * CD);
    k_count_rows<<<(CE + 255) / 256, 256, 0, stream>>>(ent_adj, CE, cnt);
    k_avg_scatter<<<(CE + 7) / 8, 256, 0, stream>>>(ent_adj, ent_adj + CE, CE, ent_emb, cnt, feat_e);
    // ---- rel_feature = avg(rel_adj, rel_emb) ----
    k_zero_f32<<<ZB, 256, 0, stream>>>(cnt, (size_t)CN);
    k_zero_f32<<<ZB, 256, 0, stream>>>(feat_r, (size_t)CN * CD);
    k_count_rows<<<(CE + 255) / 256, 256, 0, stream>>>(rel_adj, CE, cnt);
    k_avg_scatter<<<(CE + 7) / 8, 256, 0, stream>>>(rel_adj, rel_adj + CE, CE, rel_emb, cnt, feat_r);
    // ---- rels_sum (shared by both GATs) ----
    k_zero_f32<<<ZB, 256, 0, stream>>>(rels, (size_t)CT * CD);
    k_rels_scatter<<<(CT + 7) / 8, 256, 0, stream>>>(r_index, r_index + CT, r_val, CT, rel_emb, rels);
    k_l2norm_rows<<<(CT + 7) / 8, 256, 0, stream>>>(rels, rels, CT, CD);

    auto run_gat = [&](const float* feat, const float* attn, const float* gatek,
                       const float* proxy, const float* bias, int jbase) {
        size_t nf = (size_t)CN * CD;
        k_tanh_to<<<(unsigned)((nf + 255) / 256), 256, 0, stream>>>(feat, featsA, joint, jbase, nf);
        for (int l = 0; l < 2; ++l) {
            k_zero_f32<<<ZB, 256, 0, stream>>>((float*)amax, (size_t)CN);
            k_zero_f32<<<ZB, 256, 0, stream>>>(denom, (size_t)CN);
            k_zero_f32<<<ZB, 256, 0, stream>>>(featsB, nf);
            k_att_max<<<(CT + 7) / 8, 256, 0, stream>>>(rels, attn + l * CD, adj, CT, att, amax);
            k_exp_denom<<<(CT + 255) / 256, 256, 0, stream>>>(att, amax, adj, denom, CT);
            k_layer_scatter<<<(CT + 7) / 8, 256, 0, stream>>>(featsA, rels, att, denom,
                                                              adj, adj + CT, CT, featsB);
            k_tanh_to<<<(unsigned)((nf + 255) / 256), 256, 0, stream>>>(featsB, featsA, joint,
                                                                        jbase + (l + 1) * CD, nf);
        }
        // proxy attention + gate
        k_l2norm_rows<<<(CPX + 7) / 8, 256, 0, stream>>>(proxy, pnorm, CPX, CDD);
        k_proxy<<<CN, CPX, 0, stream>>>(joint, jbase, proxy, pnorm, pf, pfbf);
        k_transpose_to_bf16<<<(CDD * CDD + 255) / 256, 256, 0, stream>>>(gatek, gktbf, CDD, CDD);
        // gkout(N x 384) = pf(N x 384) @ gate_k(384 x 384)   [WMMA bf16, B pre-transposed]
        {
            int ntiles = CDD / 16;                       // 24
            dim3 grid((ntiles + 15) / 16, CN / 16);      // (2, 1875)
            k_gemm_bf16_tn<<<grid, 128, 0, stream>>>(pfbf, gktbf, gkout, CN, CDD, CDD);
        }
        size_t tot = (size_t)CN * CDD;
        k_combine<<<(unsigned)((tot + 255) / 256), 256, 0, stream>>>(gkout, bias, pf, joint,
                                                                     jbase, tot);
    };

    run_gat(feat_e, e_attn, e_gate, e_proxy, e_bias, 0);
    run_gat(feat_r, r_attn, r_gate, r_proxy, r_bias, CDD);

    // ---- align loss ----
    k_to_bf16<<<ZB, 256, 0, stream>>>(joint, jointbf, (size_t)CN * CJD);
    k_rowsq<<<(CN + 7) / 8, 256, 0, stream>>>(joint, embsq, CN, CJD);
    k_build_pairs<<<CB, 128, 0, stream>>>(joint, pairs, Abf, pos, asq, embsq);
    // ab(2048 x 30000) = Abf(2048 x 768) @ jointbf(30000 x 768)^T  [WMMA bf16, dominant 94 GFLOP]
    {
        int ntiles = CN / 16;                            // 1875
        dim3 grid((ntiles + 15) / 16, (2 * CB) / 16);    // (118, 128)
        k_gemm_bf16_tn<<<grid, 128, 0, stream>>>(Abf, jointbf, ab, 2 * CB, CN, CJD);
    }
    k_row_loss<<<2 * CB, 256, 0, stream>>>(ab, pos, asq, embsq, pairs, lse);
    k_final<<<1, 256, 0, stream>>>(lse, out);
}
